// APPNP_48369921687751
// MI455X (gfx1250) — matrix operand order repaired and checked by reading, not compiled, not run on hardware
//
#include <hip/hip_runtime.h>
#include <stdint.h>

typedef _Float16 v16h __attribute__((ext_vector_type(16)));
typedef float    v8f  __attribute__((ext_vector_type(8)));

#define NFEAT  512
#define NHID   256
#define NCLASS 64
#define KPROP  10
#define ALPHA  0.1f

#define MTILE      64                 // rows of X per block
#define LDS_STRIDE (NHID + 8)        // f16 elements per LDS row (pad vs bank conflicts)

// ---------------------------------------------------------------------------
// Pre-pass: W1T[n][k] = (f16)W1[k][n], W2T[n][k] = (f16)W2[k][n].
// Makes every WMMA B-fragment 16 contiguous f16 (two b128 loads, zero cvt).
// ---------------------------------------------------------------------------
__global__ void k_prep_w(const float* __restrict__ W1, const float* __restrict__ W2,
                         _Float16* __restrict__ W1T, _Float16* __restrict__ W2T)
{
  int i = blockIdx.x * blockDim.x + threadIdx.x;
  if (i < NFEAT * NHID) {
    int k = i / NHID, n = i % NHID;          // coalesced read of W1
    W1T[(size_t)n * NFEAT + k] = (_Float16)W1[i];
  }
  if (i < NHID * NCLASS) {
    int k = i / NCLASS, n = i % NCLASS;
    W2T[(size_t)n * NHID + k] = (_Float16)W2[i];
  }
}

// ---------------------------------------------------------------------------
// Fused GEMM: h = relu(X@W1 + b1) @ W2 + b2, written to both h0 and hc.
// 128 threads = 4 wave32; wave w owns rows [blockRow+16w, blockRow+16w+16).
// Stage 1 accumulates 16 WMMA n-tiles (256 cols) in VGPRs with a software-
// pipelined B stream (2 groups of 4 fragments in flight); stores the H1 tile
// to LDS as f16; stage 2 runs 4 WMMA n-tiles (64 cols) off the LDS tile.
// ---------------------------------------------------------------------------
__global__ __launch_bounds__(128) void appnp_gemm_fused(
    const float* __restrict__ X,
    const _Float16* __restrict__ W1T, const float* __restrict__ b1,
    const _Float16* __restrict__ W2T, const float* __restrict__ b2,
    float* __restrict__ h0, float* __restrict__ hc, int N)
{
  extern __shared__ _Float16 H1[];          // [MTILE][LDS_STRIDE] f16

  const int lane = threadIdx.x & 31;
  const int wave = threadIdx.x >> 5;
  const int l15  = lane & 15;
  const int hi   = lane >> 4;               // 0: lanes 0-15, 1: lanes 16-31

  const int blockRow = blockIdx.x * MTILE;
  int rowA = blockRow + wave * 16 + l15;
  const int rowC = rowA < N ? rowA : (N - 1);   // clamp loads; stores guarded
  const float* xrow = X + (size_t)rowC * NFEAT;

  // ---------------- stage 1: H1 = relu(X @ W1 + b1) ----------------
  v8f acc[16] = {};
  for (int k = 0; k < NFEAT; k += 32) {
    // A fragment (16x32 f16): lane<16 holds row M=l15, K = k..k+15;
    // lanes 16-31 hold K = k+16..k+31 (matches B's contiguous K order).
    const float4* xq = (const float4*)(xrow + k + hi * 16);   // 32B aligned
    float4 p0 = xq[0];
    float4 p1 = xq[1];
    float4 p2 = xq[2];
    float4 p3 = xq[3];
    v16h a;
    a[0]  = (_Float16)p0.x; a[1]  = (_Float16)p0.y;
    a[2]  = (_Float16)p0.z; a[3]  = (_Float16)p0.w;
    a[4]  = (_Float16)p1.x; a[5]  = (_Float16)p1.y;
    a[6]  = (_Float16)p1.z; a[7]  = (_Float16)p1.w;
    a[8]  = (_Float16)p2.x; a[9]  = (_Float16)p2.y;
    a[10] = (_Float16)p2.z; a[11] = (_Float16)p2.w;
    a[12] = (_Float16)p3.x; a[13] = (_Float16)p3.y;
    a[14] = (_Float16)p3.z; a[15] = (_Float16)p3.w;

    const int kb = k + hi * 16;   // B: lanes<16 K=k..k+15, lanes>=16 K=k+16..k+31
    // per-lane column base: this lane's column l15, K offset kb
    const _Float16* Wbase = W1T + (size_t)l15 * NFEAT + kb;
    // B fragment for n-tile nt lives at a compile-time byte offset nt*16*NFEAT*2
#define LOADB1(nt) (*(const v16h*)(Wbase + (size_t)(nt) * 16 * NFEAT))

    // software pipeline: 4 B-fragments per group, next group's loads issued
    // before the current group's WMMAs consume theirs.
    v16h bq0 = LOADB1(0), bq1 = LOADB1(1), bq2 = LOADB1(2), bq3 = LOADB1(3);
#pragma unroll
    for (int g = 0; g < 4; ++g) {
      v16h c0 = bq0, c1 = bq1, c2 = bq2, c3 = bq3;
      if (g < 3) {
        c0 = LOADB1(4 * g + 4);
        c1 = LOADB1(4 * g + 5);
        c2 = LOADB1(4 * g + 6);
        c3 = LOADB1(4 * g + 7);
      }
      acc[4 * g + 0] = __builtin_amdgcn_wmma_f32_16x16x32_f16(
          false, a, false, bq0, (short)0, acc[4 * g + 0], false, false);
      acc[4 * g + 1] = __builtin_amdgcn_wmma_f32_16x16x32_f16(
          false, a, false, bq1, (short)0, acc[4 * g + 1], false, false);
      acc[4 * g + 2] = __builtin_amdgcn_wmma_f32_16x16x32_f16(
          false, a, false, bq2, (short)0, acc[4 * g + 2], false, false);
      acc[4 * g + 3] = __builtin_amdgcn_wmma_f32_16x16x32_f16(
          false, a, false, bq3, (short)0, acc[4 * g + 3], false, false);
      bq0 = c0; bq1 = c1; bq2 = c2; bq3 = c3;
    }
#undef LOADB1
  }
  // epilogue: bias + relu -> LDS (f16).  C layout: VGPR r -> M = r + 8*hi, N = l15.
#pragma unroll
  for (int nt = 0; nt < 16; ++nt) {
    const int n = nt * 16 + l15;
    const float bias = b1[n];
#pragma unroll
    for (int r = 0; r < 8; ++r) {
      const int m = wave * 16 + r + hi * 8;
      float v = acc[nt][r] + bias;
      H1[m * LDS_STRIDE + n] = (_Float16)(v > 0.0f ? v : 0.0f);
    }
  }
  __syncthreads();

  // ---------------- stage 2: H = H1 @ W2 + b2 ----------------
  v8f acc2[4] = {};
  for (int k = 0; k < NHID; k += 32) {
    v16h a;
    const _Float16* hp = &H1[(wave * 16 + l15) * LDS_STRIDE + k + hi * 16];
#pragma unroll
    for (int j = 0; j < 8; ++j) {
      a[j]     = hp[j];
      a[j + 8] = hp[j + 8];
    }
    const int kb = k + hi * 16;
    const _Float16* Wbase2 = W2T + (size_t)l15 * NHID + kb;
    // all 4 B fragments issued before any WMMA consumes them
    v16h b0 = *(const v16h*)(Wbase2 + (size_t)0 * 16 * NHID);
    v16h b1v = *(const v16h*)(Wbase2 + (size_t)1 * 16 * NHID);
    v16h b2v = *(const v16h*)(Wbase2 + (size_t)2 * 16 * NHID);
    v16h b3v = *(const v16h*)(Wbase2 + (size_t)3 * 16 * NHID);
    acc2[0] = __builtin_amdgcn_wmma_f32_16x16x32_f16(
        false, a, false, b0, (short)0, acc2[0], false, false);
    acc2[1] = __builtin_amdgcn_wmma_f32_16x16x32_f16(
        false, a, false, b1v, (short)0, acc2[1], false, false);
    acc2[2] = __builtin_amdgcn_wmma_f32_16x16x32_f16(
        false, a, false, b2v, (short)0, acc2[2], false, false);
    acc2[3] = __builtin_amdgcn_wmma_f32_16x16x32_f16(
        false, a, false, b3v, (short)0, acc2[3], false, false);
  }
#pragma unroll
  for (int nt = 0; nt < 4; ++nt) {
    const int n = nt * 16 + l15;
    const float bias = b2[n];
#pragma unroll
    for (int r = 0; r < 8; ++r) {
      const int row = blockRow + wave * 16 + r + hi * 8;
      if (row < N) {
        float v = acc2[nt][r] + bias;
        h0[(size_t)row * NCLASS + n] = v;
        hc[(size_t)row * NCLASS + n] = v;
      }
    }
  }
}

// ---------------------------------------------------------------------------
// GCN norm precompute: deg[i] = 1 (self loop) + indeg(i); dinv = rsqrt(deg)
// ---------------------------------------------------------------------------
__global__ void k_deg_init(float* __restrict__ deg, int N)
{
  int i = blockIdx.x * blockDim.x + threadIdx.x;
  if (i < N) deg[i] = 1.0f;
}

__global__ void k_deg_count(const long long* __restrict__ dst,
                            float* __restrict__ deg, int E)
{
  int e = blockIdx.x * blockDim.x + threadIdx.x;
  if (e < E) atomicAdd(&deg[(int)dst[e]], 1.0f);
}

__global__ void k_dinv(const float* __restrict__ deg, float* __restrict__ dinv, int N)
{
  int i = blockIdx.x * blockDim.x + threadIdx.x;
  if (i < N) dinv[i] = rsqrtf(deg[i]);   // deg >= 1 always
}

// ---------------------------------------------------------------------------
// Propagation step kernels. Working set (~110 MB) is L2-resident.
// ---------------------------------------------------------------------------
// agg[i,:] = dinv[i]^2 * hc[i,:]   (self-loop term); one float4 per thread
__global__ void k_selfinit(const float* __restrict__ hc,
                           const float* __restrict__ dinv,
                           float* __restrict__ agg, int N)
{
  int i = blockIdx.x * blockDim.x + threadIdx.x;     // N * 16 float4's
  if (i >= N * (NCLASS / 4)) return;
  int node = i >> 4;
  float d = dinv[node];
  float w = d * d;
  float4 v = ((const float4*)hc)[i];
  float4 o;
  o.x = w * v.x; o.y = w * v.y; o.z = w * v.z; o.w = w * v.w;
  ((float4*)agg)[i] = o;
}

// one wave32 per edge; lane L covers features 2L, 2L+1
__global__ __launch_bounds__(256) void k_scatter(
    const long long* __restrict__ src, const long long* __restrict__ dst,
    const float* __restrict__ dinv, const float* __restrict__ hc,
    float* __restrict__ agg, int E)
{
  int gid  = blockIdx.x * blockDim.x + threadIdx.x;
  int e    = gid >> 5;
  int lane = gid & 31;
  if (e >= E) return;
  int s = (int)src[e];
  int d = (int)dst[e];
  float w = dinv[s] * dinv[d];
  float2 v = ((const float2*)(hc + (size_t)s * NCLASS))[lane];
  float* ap = agg + (size_t)d * NCLASS + 2 * lane;
  atomicAdd(ap + 0, w * v.x);
  atomicAdd(ap + 1, w * v.y);
}

// hc = (1-alpha)*agg + alpha*h0
__global__ void k_combine(const float* __restrict__ agg,
                          const float* __restrict__ h0,
                          float* __restrict__ hc, int N)
{
  int i = blockIdx.x * blockDim.x + threadIdx.x;
  if (i >= N * (NCLASS / 4)) return;
  float4 a = ((const float4*)agg)[i];
  float4 z = ((const float4*)h0)[i];
  float4 o;
  o.x = (1.0f - ALPHA) * a.x + ALPHA * z.x;
  o.y = (1.0f - ALPHA) * a.y + ALPHA * z.y;
  o.z = (1.0f - ALPHA) * a.z + ALPHA * z.z;
  o.w = (1.0f - ALPHA) * a.w + ALPHA * z.w;
  ((float4*)hc)[i] = o;
}

// ---------------------------------------------------------------------------
extern "C" void kernel_launch(void* const* d_in, const int* in_sizes, int n_in,
                              void* d_out, int out_size, void* d_ws, size_t ws_size,
                              hipStream_t stream)
{
  const float*     X   = (const float*)d_in[0];
  const long long* EI  = (const long long*)d_in[1];   // int64 [2, E]
  const float*     W1  = (const float*)d_in[2];
  const float*     b1  = (const float*)d_in[3];
  const float*     W2  = (const float*)d_in[4];
  const float*     b2  = (const float*)d_in[5];

  const int N = in_sizes[0] / NFEAT;
  const int E = in_sizes[1] / 2;
  const long long* src = EI;
  const long long* dst = EI + E;

  float* hc  = (float*)d_out;                         // [N, NCLASS] current h
  float* h0  = (float*)d_ws;                          // [N, NCLASS]
  float* agg = h0 + (size_t)N * NCLASS;               // [N, NCLASS]
  float* deg = agg + (size_t)N * NCLASS;              // [N]
  float* dnv = deg + N;                               // [N]
  // f16 weight tables, 64B-aligned
  uintptr_t p = (uintptr_t)(dnv + N);
  p = (p + 63u) & ~(uintptr_t)63u;
  _Float16* W1T = (_Float16*)p;                       // [NHID][NFEAT]
  _Float16* W2T = W1T + (size_t)NHID * NFEAT;         // [NCLASS][NHID]

  // weight transpose+convert (tiny, stays hot in L2 for all GEMM blocks)
  k_prep_w<<<(NFEAT * NHID + 255) / 256, 256, 0, stream>>>(W1, W2, W1T, W2T);

  // fused MLP (both GEMMs) -> h0 and hc
  {
    int blocks = (N + MTILE - 1) / MTILE;
    size_t lds = (size_t)MTILE * LDS_STRIDE * sizeof(_Float16);
    appnp_gemm_fused<<<blocks, 128, lds, stream>>>(X, W1T, b1, W2T, b2, h0, hc, N);
  }

  // symmetric normalization
  k_deg_init<<<(N + 255) / 256, 256, 0, stream>>>(deg, N);
  k_deg_count<<<(E + 255) / 256, 256, 0, stream>>>(dst, deg, E);
  k_dinv<<<(N + 255) / 256, 256, 0, stream>>>(deg, dnv, N);

  // K propagation rounds
  const int nf4 = N * (NCLASS / 4);
  for (int it = 0; it < KPROP; ++it) {
    k_selfinit<<<(nf4 + 255) / 256, 256, 0, stream>>>(hc, dnv, agg, N);
    k_scatter<<<(E + 7) / 8, 256, 0, stream>>>(src, dst, dnv, hc, agg, E);
    k_combine<<<(nf4 + 255) / 256, 256, 0, stream>>>(agg, h0, hc, N);
  }
}
